// SENT_RNN_36326833390152
// MI455X (gfx1250) — matrix operand order repaired and checked by reading, not compiled, not run on hardware
//
#include <hip/hip_runtime.h>

// ---------------------------------------------------------------------------
// SummaRuNNer-style sent-RNN for MI455X (gfx1250).
// All GEMMs use v_wmma_f32_16x16x32_bf16 (bf16 A/B, f32 accumulate).
// GEMM inner loop is software-pipelined with an even/odd register-set unroll
// (no rotation copies, no spills): the next 32-K chunk's loads issue while
// the current chunk's WMMAs execute. Requires K % 64 == 0 (K = 256 or 512).
// Wave32 everywhere; shuffles/reductions written for warpSize==32.
// ---------------------------------------------------------------------------

#define BD   512          // docs
#define LD   100          // max doc len
#define DD   256          // emb dim
#define HD   256          // hidden
#define G3   768          // 3*H
#define TDW  512          // 2*H BiGRU width
#define ROWS (BD*LD)      // 51200

typedef __bf16 bf16;
typedef bf16  bf16x8 __attribute__((ext_vector_type(8)));
typedef bf16  v16bf  __attribute__((ext_vector_type(16)));
typedef float v8f    __attribute__((ext_vector_type(8)));

static __device__ __forceinline__ float sigmoidf_(float x) {
  return 1.0f / (1.0f + __expf(-x));
}

static __device__ __forceinline__ v16bf load16bf(const bf16* p0, const bf16* p1) {
  bf16x8 a = *(const bf16x8*)p0;   // 16-byte global_load_b128
  bf16x8 b = *(const bf16x8*)p1;
  return __builtin_shufflevector(a, b, 0,1,2,3,4,5,6,7,8,9,10,11,12,13,14,15);
}

// One wave computes a 16(M) x 64(N) tile of C = A @ W^T.
// A: row-major [M,K] bf16 with row stride lda. W: row-major [N,K] bf16, row stride ldw.
// Per-lane operand striping uses contiguous K, same order for A and B (16-bit A 16x32, B 32x16):
//   A fragment: lane row m=l16 holds 16 contiguous K at kk+half*16.
//   B fragment: lane column n=l16 holds 16 contiguous K at kk+half*16.
static __device__ __forceinline__ void wmma_tile_16x64(
    const bf16* __restrict__ A, int lda,
    const bf16* __restrict__ W, int ldw,
    float* __restrict__ C, int ldc,
    int m0, int n0, int K)
{
  const int lane = threadIdx.x & 31;
  const int half = lane >> 4;   // 0: lanes 0-15, 1: lanes 16-31
  const int l16  = lane & 15;

  v8f acc[4] = { v8f{}, v8f{}, v8f{}, v8f{} };

  const bf16* arow = A + (size_t)(m0 + l16) * lda;
  const bf16* wrow[4];
#pragma unroll
  for (int t = 0; t < 4; ++t)
    wrow[t] = W + (size_t)(n0 + t * 16 + l16) * ldw + half * 16;

  // Even/odd pipelined K loop: set 0 covers kk, set 1 covers kk+32.
  v16bf a0, b0[4], a1, b1[4];
  a0 = load16bf(arow + half * 16, arow + half * 16 + 8);
#pragma unroll
  for (int t = 0; t < 4; ++t) b0[t] = load16bf(wrow[t], wrow[t] + 8);

  for (int kk = 0; kk < K; kk += 64) {
    // Issue odd-chunk loads, then compute even chunk.
    __builtin_prefetch(arow + kk + 256, 0, 1);           // global_prefetch_b8
    a1 = load16bf(arow + kk + 32 + half * 16, arow + kk + 40 + half * 16);
#pragma unroll
    for (int t = 0; t < 4; ++t)
      b1[t] = load16bf(wrow[t] + kk + 32, wrow[t] + kk + 40);
#pragma unroll
    for (int t = 0; t < 4; ++t)
      acc[t] = __builtin_amdgcn_wmma_f32_16x16x32_bf16(
          false, a0, false, b0[t], (short)0, acc[t], false, false);

    // Issue next even-chunk loads, then compute odd chunk.
    if (kk + 64 < K) {
      a0 = load16bf(arow + kk + 64 + half * 16, arow + kk + 72 + half * 16);
#pragma unroll
      for (int t = 0; t < 4; ++t)
        b0[t] = load16bf(wrow[t] + kk + 64, wrow[t] + kk + 72);
    }
#pragma unroll
    for (int t = 0; t < 4; ++t)
      acc[t] = __builtin_amdgcn_wmma_f32_16x16x32_bf16(
          false, a1, false, b1[t], (short)0, acc[t], false, false);
  }

  // C 16x16 f32 layout: VGPR i holds (m0 + half*8 + i, n0 + t*16 + l16)
#pragma unroll
  for (int t = 0; t < 4; ++t) {
    float* crow = C + (size_t)(m0 + half * 8) * ldc + n0 + t * 16 + l16;
#pragma unroll
    for (int i = 0; i < 8; ++i) crow[(size_t)i * ldc] = acc[t][i];
  }
}

// ---------------------------------------------------------------------------
// Generic WMMA GEMM: block = 128 thr = 4 waves, each wave a 16x64 tile.
// grid.x = M/16, grid.y = N/256.
__global__ void gemm_bf16_f32(const bf16* __restrict__ A, int lda,
                              const bf16* __restrict__ W, int ldw,
                              float* __restrict__ C, int ldc, int K)
{
  int wave = threadIdx.x >> 5;
  int m0 = blockIdx.x * 16;
  int n0 = blockIdx.y * 256 + wave * 64;
  wmma_tile_16x64(A, lda, W, ldw, C, ldc, m0, n0, K);
}

// GRU step GEMMs: grid.z = 4 -> (dir, part). part 0: h_prev @ W_hh^T,
// part 1: x_t @ W_ih^T. Both dirs in one launch per timestep.
__global__ void gru_step_gemm(const bf16* __restrict__ xpad,   // [B,L,D] bf16
                              const bf16* __restrict__ hbf,    // [2,B,H] bf16
                              const bf16* __restrict__ whh,    // [2,3H,H]
                              const bf16* __restrict__ wih,    // [2,3H,D]
                              float* __restrict__ gh,          // [4,B,3H]
                              int s)
{
  int z    = blockIdx.z;
  int dir  = z >> 1;
  int part = z & 1;
  const bf16* A; int lda; const bf16* W;
  if (part == 0) { A = hbf + (size_t)dir * BD * HD; lda = HD; W = whh + (size_t)dir * G3 * HD; }
  else {
    int t = dir ? (LD - 1 - s) : s;
    A = xpad + (size_t)t * DD; lda = LD * DD;        // row b lives at (b*L + t)*D
    W = wih + (size_t)dir * G3 * DD;
  }
  float* C = gh + (size_t)z * BD * G3;
  int wave = threadIdx.x >> 5;
  int m0 = blockIdx.x * 16;
  int n0 = blockIdx.y * 256 + wave * 64;
  wmma_tile_16x64(A, lda, W, G3, C, G3, m0, n0, HD);
}

// Fused GRU gates for both directions: h2 = (1-z)*n + z*h.
__global__ void gru_gate(const float* __restrict__ gh, float* __restrict__ hprev,
                         bf16* __restrict__ hbf, float* __restrict__ hlb,
                         bf16* __restrict__ hlbb,
                         const float* __restrict__ bihf, const float* __restrict__ bhhf,
                         const float* __restrict__ bihb, const float* __restrict__ bhhb,
                         int s)
{
  int idx = blockIdx.x * blockDim.x + threadIdx.x;       // [0, 2*B*H)
  if (idx >= 2 * BD * HD) return;
  int dir = idx >> 17;                 // B*H = 131072 = 2^17
  int rem = idx & (BD * HD - 1);
  int b = rem >> 8, j = rem & 255;
  const float* ghh = gh + (size_t)(dir * 2 + 0) * BD * G3 + (size_t)b * G3;
  const float* ghx = gh + (size_t)(dir * 2 + 1) * BD * G3 + (size_t)b * G3;
  const float* bi = dir ? bihb : bihf;
  const float* bh = dir ? bhhb : bhhf;
  float ir = ghx[j]       + bi[j];
  float iz = ghx[j + 256] + bi[j + 256];
  float in = ghx[j + 512] + bi[j + 512];
  float hr = ghh[j]       + bh[j];
  float hz = ghh[j + 256] + bh[j + 256];
  float hn = ghh[j + 512] + bh[j + 512];
  float r = sigmoidf_(ir + hr);
  float zg = sigmoidf_(iz + hz);
  float n = tanhf(in + r * hn);
  float h = hprev[idx];
  float h2 = (1.0f - zg) * n + zg * h;
  hprev[idx] = h2;
  hbf[idx] = (bf16)h2;
  int t = dir ? (LD - 1 - s) : s;
  size_t o = ((size_t)t * BD + b) * TDW + dir * HD + j;  // hlb[t][b][dir*H + j]
  hlb[o] = h2;
  hlbb[o] = (bf16)h2;
}

// ---------------------------------------------------------------------------
__global__ void prefix_lens(const int* __restrict__ dl, int* __restrict__ off) {
  if (threadIdx.x == 0 && blockIdx.x == 0) {
    int acc = 0;
    for (int i = 0; i < BD; ++i) { off[i] = acc; acc += dl[i]; }
  }
}

__global__ void cvt_f32_bf16(const float* __restrict__ s, bf16* __restrict__ d, int n) {
  int i = blockIdx.x * blockDim.x + threadIdx.x;
  if (i < n) d[i] = (bf16)s[i];
}

__global__ void scatter_pad(const float* __restrict__ sents, const int* __restrict__ off,
                            const int* __restrict__ dl, bf16* __restrict__ xpad) {
  int idx = blockIdx.x * blockDim.x + threadIdx.x;      // B*L*D
  if (idx >= BD * LD * DD) return;
  int b = idx / (LD * DD);
  int rem = idx - b * (LD * DD);
  int l = rem / DD, d = rem - l * DD;
  float v = 0.0f;
  if (l < dl[b]) v = sents[((size_t)off[b] + l) * DD + d];
  xpad[idx] = (bf16)v;
}

__global__ void init_h(float* __restrict__ hprev, bf16* __restrict__ hbf) {
  int i = blockIdx.x * blockDim.x + threadIdx.x;
  if (i < 2 * BD * HD) { hprev[i] = 0.0f; hbf[i] = (bf16)0.0f; }
}

// Masked mean over valid positions -> docs_raw [B, TD] (f32 + bf16).
__global__ void pool_docs(const float* __restrict__ hlb, const int* __restrict__ dl,
                          float* __restrict__ draw, bf16* __restrict__ drawb) {
  int idx = blockIdx.x * blockDim.x + threadIdx.x;      // B*TD
  if (idx >= BD * TDW) return;
  int b = idx >> 9, e = idx & 511;
  int n = dl[b];
  float s = 0.0f;
  for (int l = 0; l < n; ++l) s += hlb[((size_t)l * BD + b) * TDW + e];
  float v = s / (float)(n > 0 ? n : 1);
  draw[idx] = v; drawb[idx] = (bf16)v;
}

__global__ void docs_act(const float* __restrict__ lin, const float* __restrict__ bias,
                         bf16* __restrict__ docsb) {
  int idx = blockIdx.x * blockDim.x + threadIdx.x;
  if (idx >= BD * TDW) return;
  float v = tanhf(lin[idx] + bias[idx & 511]);
  docsb[idx] = (bf16)v;
}

__global__ void sd_add_content(float* __restrict__ sd, const float* __restrict__ wc) {
  int idx = blockIdx.x * blockDim.x + threadIdx.x;
  if (idx < BD * TDW) sd[idx] += wc[idx & 511];
}

// posb[l] = abs_emb[l] @ Wabs + rel_emb[round(l/10)] @ Wrel + bias
__global__ void pos_bias(const float* __restrict__ abse, const float* __restrict__ rele,
                         const float* __restrict__ wabs, const float* __restrict__ wrel,
                         const float* __restrict__ bias, float* __restrict__ posb) {
  int l = blockIdx.x * blockDim.x + threadIdx.x;
  if (l >= LD) return;
  int ridx = (l + 5) / 10; if (ridx > 10) ridx = 10;
  float a = 0.0f, r = 0.0f;
  for (int k = 0; k < 100; ++k) {
    a += abse[l * 100 + k] * wabs[k];
    r += rele[ridx * 100 + k] * wrel[k];
  }
  posb[l] = a + r + bias[0];
}

// base[l*B+b] = <hlb[l,b,:], sd[b,:]+Wcontent> + posb[l]; one wave per (l,b) dot.
__global__ void base_scores(const float* __restrict__ hlb, const float* __restrict__ sd,
                            const float* __restrict__ posb, float* __restrict__ base) {
  int wave = threadIdx.x >> 5, lane = threadIdx.x & 31;
  int idx = blockIdx.x * 8 + wave;                      // [0, L*B)
  if (idx >= LD * BD) return;
  int l = idx >> 9, b = idx & 511;
  const float* h = hlb + (size_t)idx * TDW;
  const float* v = sd + (size_t)b * TDW;
  float s = 0.0f;
#pragma unroll
  for (int j = 0; j < 16; ++j) { int e = lane + 32 * j; s += h[e] * v[e]; }
#pragma unroll
  for (int m = 16; m >= 1; m >>= 1) s += __shfl_xor(s, m, 32);
  if (lane == 0) base[idx] = s + posb[l];
}

// Persistent novelty scan: one wave per doc, ses state in VGPRs (16 f32/lane).
__global__ void novelty_scan(const float* __restrict__ hlb, const float* __restrict__ xw,
                             const float* __restrict__ base, float* __restrict__ probs) {
  int wave = threadIdx.x >> 5, lane = threadIdx.x & 31;
  int b = blockIdx.x * 8 + wave;
  if (b >= BD) return;
  float ses[16];
#pragma unroll
  for (int j = 0; j < 16; ++j) ses[j] = 0.0f;
  for (int l = 0; l < LD; ++l) {
    size_t row = ((size_t)l * BD + b) * TDW;
    __builtin_prefetch(xw + row + BD * TDW + lane, 0, 1);
    float s = 0.0f;
#pragma unroll
    for (int j = 0; j < 16; ++j) s += xw[row + lane + 32 * j] * tanhf(ses[j]);
#pragma unroll
    for (int m = 16; m >= 1; m >>= 1) s += __shfl_xor(s, m, 32);
    float prob = sigmoidf_(base[l * BD + b] - s);        // nov = -s
#pragma unroll
    for (int j = 0; j < 16; ++j) ses[j] += hlb[row + lane + 32 * j] * prob;
    if (lane == 0) probs[l * BD + b] = prob;
  }
}

__global__ void gather_out(const float* __restrict__ probs, const int* __restrict__ off,
                           const int* __restrict__ dl, float* __restrict__ out) {
  int b = blockIdx.x;
  int n = dl[b], o = off[b];
  for (int p = threadIdx.x; p < n; p += blockDim.x)
    out[o + p] = probs[p * BD + b];
}

// ---------------------------------------------------------------------------
extern "C" void kernel_launch(void* const* d_in, const int* in_sizes, int n_in,
                              void* d_out, int out_size, void* d_ws, size_t ws_size,
                              hipStream_t stream) {
  (void)in_sizes; (void)n_in; (void)out_size; (void)ws_size;
  const float* sents = (const float*)d_in[0];
  const float* wihf  = (const float*)d_in[1];
  const float* whhf  = (const float*)d_in[2];
  const float* bihf  = (const float*)d_in[3];
  const float* bhhf  = (const float*)d_in[4];
  const float* wihb  = (const float*)d_in[5];
  const float* whhb  = (const float*)d_in[6];
  const float* bihb  = (const float*)d_in[7];
  const float* bhhb  = (const float*)d_in[8];
  const float* abse  = (const float*)d_in[9];
  const float* rele  = (const float*)d_in[10];
  const float* wdocw = (const float*)d_in[11];
  const float* wdocb = (const float*)d_in[12];
  const float* wcont = (const float*)d_in[13];
  const float* wsal  = (const float*)d_in[14];
  const float* wnov  = (const float*)d_in[15];
  const float* wabs  = (const float*)d_in[16];
  const float* wrel  = (const float*)d_in[17];
  const float* biasp = (const float*)d_in[18];
  const int*   dlens = (const int*)d_in[19];
  float* out = (float*)d_out;

  char* w = (char*)d_ws; size_t o = 0;
  auto alloc = [&](size_t bytes) -> void* {
    void* p = w + o; o = (o + bytes + 255) & ~(size_t)255; return p;
  };
  int*   off      = (int*)  alloc(BD * 4);
  bf16*  xpad     = (bf16*) alloc((size_t)BD * LD * DD * 2);
  bf16*  wih_b16  = (bf16*) alloc((size_t)2 * G3 * DD * 2);
  bf16*  whh_b16  = (bf16*) alloc((size_t)2 * G3 * HD * 2);
  bf16*  wdoc_b16 = (bf16*) alloc((size_t)TDW * TDW * 2);
  bf16*  wsal_b16 = (bf16*) alloc((size_t)TDW * TDW * 2);
  bf16*  wnov_b16 = (bf16*) alloc((size_t)TDW * TDW * 2);
  float* hprev    = (float*)alloc((size_t)2 * BD * HD * 4);
  bf16*  hbf      = (bf16*) alloc((size_t)2 * BD * HD * 2);
  float* gh       = (float*)alloc((size_t)4 * BD * G3 * 4);
  float* hlb      = (float*)alloc((size_t)ROWS * TDW * 4);
  bf16*  hlbb     = (bf16*) alloc((size_t)ROWS * TDW * 2);
  float* draw     = (float*)alloc((size_t)BD * TDW * 4);
  bf16*  drawb    = (bf16*) alloc((size_t)BD * TDW * 2);
  float* dlin     = (float*)alloc((size_t)BD * TDW * 4);
  bf16*  docsb    = (bf16*) alloc((size_t)BD * TDW * 2);
  float* sd       = (float*)alloc((size_t)BD * TDW * 4);
  float* posb     = (float*)alloc(128 * 4);
  float* baseb    = (float*)alloc((size_t)ROWS * 4);
  float* xw       = (float*)alloc((size_t)(ROWS + BD) * TDW * 4); // +1 row pad for prefetch
  float* probs    = (float*)alloc((size_t)ROWS * 4);

  // 0) prefix offsets of doc lengths
  prefix_lens<<<1, 32, 0, stream>>>(dlens, off);

  // 1) weights -> bf16
  int nW = G3 * DD;
  cvt_f32_bf16<<<(nW + 255) / 256, 256, 0, stream>>>(wihf, wih_b16, nW);
  cvt_f32_bf16<<<(nW + 255) / 256, 256, 0, stream>>>(wihb, wih_b16 + nW, nW);
  cvt_f32_bf16<<<(nW + 255) / 256, 256, 0, stream>>>(whhf, whh_b16, nW);
  cvt_f32_bf16<<<(nW + 255) / 256, 256, 0, stream>>>(whhb, whh_b16 + nW, nW);
  int nT = TDW * TDW;
  cvt_f32_bf16<<<(nT + 255) / 256, 256, 0, stream>>>(wdocw, wdoc_b16, nT);
  cvt_f32_bf16<<<(nT + 255) / 256, 256, 0, stream>>>(wsal, wsal_b16, nT);
  cvt_f32_bf16<<<(nT + 255) / 256, 256, 0, stream>>>(wnov, wnov_b16, nT);

  // 2) zero-padded bf16 activations, zero hidden state
  scatter_pad<<<(BD * LD * DD + 255) / 256, 256, 0, stream>>>(sents, off, dlens, xpad);
  init_h<<<(2 * BD * HD + 255) / 256, 256, 0, stream>>>(hprev, hbf);

  // 3) BiGRU: 100 timesteps, both directions + both GEMM parts per launch
  dim3 sgrid(BD / 16, G3 / 256, 4);            // 32 x 3 x 4 blocks of 128
  for (int s = 0; s < LD; ++s) {
    gru_step_gemm<<<sgrid, 128, 0, stream>>>(xpad, hbf, whh_b16, wih_b16, gh, s);
    gru_gate<<<(2 * BD * HD + 255) / 256, 256, 0, stream>>>(
        gh, hprev, hbf, hlb, hlbb, bihf, bhhf, bihb, bhhb, s);
  }

  // 4) doc representation
  pool_docs<<<(BD * TDW + 255) / 256, 256, 0, stream>>>(hlb, dlens, draw, drawb);
  gemm_bf16_f32<<<dim3(BD / 16, TDW / 256), 128, 0, stream>>>(
      drawb, TDW, wdoc_b16, TDW, dlin, TDW, TDW);
  docs_act<<<(BD * TDW + 255) / 256, 256, 0, stream>>>(dlin, wdocb, docsb);

  // 5) salience vector sd = docs @ Wsal^T, then fold in Wcontent
  gemm_bf16_f32<<<dim3(BD / 16, TDW / 256), 128, 0, stream>>>(
      docsb, TDW, wsal_b16, TDW, sd, TDW, TDW);
  sd_add_content<<<(BD * TDW + 255) / 256, 256, 0, stream>>>(sd, wcont);

  // 6) position biases + base scores
  pos_bias<<<1, 128, 0, stream>>>(abse, rele, wabs, wrel, biasp, posb);
  base_scores<<<ROWS / 8, 256, 0, stream>>>(hlb, sd, posb, baseb);

  // 7) novelty left factor xw = hlb @ Wnov^T (largest GEMM: 51200x512x512);
  //    Wnov (512 KB bf16) stays L2-resident, so 16-row tiles are fine.
  gemm_bf16_f32<<<dim3(ROWS / 16, TDW / 256), 128, 0, stream>>>(
      hlbb, TDW, wnov_b16, TDW, xw, TDW, TDW);

  // 8) sequential novelty scan (one wave per doc, state in VGPRs)
  novelty_scan<<<BD / 8, 256, 0, stream>>>(hlb, xw, baseb, probs);

  // 9) gather valid positions
  gather_out<<<BD, 128, 0, stream>>>(probs, off, dlens, out);
}
